// Encoder_17987323036469
// MI455X (gfx1250) — matrix operand order repaired and checked by reading, not compiled, not run on hardware
//
#include <hip/hip_runtime.h>
#include <math.h>

typedef __attribute__((ext_vector_type(16))) __bf16 bf16x16;
typedef __attribute__((ext_vector_type(8)))  __bf16 bf16x8;
typedef __attribute__((ext_vector_type(8)))  float  f32x8;

namespace {
constexpr int kB  = 64;
constexpr int kT  = 256;
constexpr int kE  = 256;
constexpr int kU  = 1024;
constexpr int kN3 = 3 * kU;   // 3072 gate columns [z | r | h]
}

// ---------------------------------------------------------------------------
// Device helpers
// ---------------------------------------------------------------------------

__device__ __forceinline__ float fast_sigmoid(float v) {
  return 1.0f / (1.0f + __expf(-v));
}

// Build a 16-element bf16 fragment from two contiguous 8-element halves.
__device__ __forceinline__ bf16x16 make_frag(const __bf16* p0, const __bf16* p1) {
  bf16x8 lo = *reinterpret_cast<const bf16x8*>(p0);
  bf16x8 hi = *reinterpret_cast<const bf16x8*>(p1);
  return __builtin_shufflevector(lo, hi, 0, 1, 2, 3, 4, 5, 6, 7,
                                 8, 9, 10, 11, 12, 13, 14, 15);
}

// ---------------------------------------------------------------------------
// Prep kernels
// ---------------------------------------------------------------------------

// dst[n*K + k] = bf16(src[k*N + n]) : convert f32 weights to bf16, transposed
// so that a WMMA B-fragment (fixed output column n, contiguous K) is a
// contiguous 32B load per lane.
__global__ void convert_transpose_kernel(const float* __restrict__ src,
                                         __bf16* __restrict__ dst,
                                         int K, int N) {
  long i = (long)blockIdx.x * blockDim.x + threadIdx.x;
  long total = (long)K * N;
  if (i >= total) return;
  int k = (int)(i / N);
  int n = (int)(i % N);
  dst[(long)n * K + k] = (__bf16)src[i];
}

// xe[b,t,k] = bf16(emb[x[b,t], k])  (embedding gather + downconvert)
__global__ void embed_bf16_kernel(const int* __restrict__ x,
                                  const float* __restrict__ emb,
                                  __bf16* __restrict__ xe) {
  long i = (long)blockIdx.x * blockDim.x + threadIdx.x;
  if (i >= (long)kB * kT * kE) return;
  int bt = (int)(i >> 8);       // kE == 256
  int k  = (int)(i & (kE - 1));
  int tok = x[bt];
  xe[i] = (__bf16)emb[(long)tok * kE + k];
}

__global__ void init_h_kernel(const float* __restrict__ h0,
                              float* __restrict__ hf,
                              __bf16* __restrict__ hbf) {
  int i = blockIdx.x * blockDim.x + threadIdx.x;
  if (i >= kB * kU) return;
  float v = h0[i];
  hf[i]  = v;
  hbf[i] = (__bf16)v;
}

__global__ void copy_state_kernel(const float* __restrict__ hf,
                                  float* __restrict__ dst) {
  int i = blockIdx.x * blockDim.x + threadIdx.x;
  if (i >= kB * kU) return;
  dst[i] = hf[i];
}

// ---------------------------------------------------------------------------
// Fused GRU step: one wave computes a 16(batch) x 16(hidden-unit) tile.
//   acc_z, acc_r accumulate BOTH h@U and xe@W contributions (K concat sums).
//   acc_rh (h@U only) and acc_xh (xe@W only) stay separate for reset_after.
// Block = 128 threads = 4 waves covering all 64 batch rows; grid.x = U/16.
// ---------------------------------------------------------------------------

__global__ void __launch_bounds__(128)
gru_step_kernel(const int*    __restrict__ x,       // [B,T] token ids
                const float*  __restrict__ bias,    // [2,3U]
                const __bf16* __restrict__ Ut,      // [3U,U]  (transposed bf16)
                const __bf16* __restrict__ Wt,      // [3U,E]  (transposed bf16)
                const __bf16* __restrict__ xe,      // [B,T,E] bf16
                const __bf16* __restrict__ hbf_in,  // [B,U]
                const float*  __restrict__ hf_in,   // [B,U]
                float*        __restrict__ out,     // [B,T,U]
                float*        __restrict__ hf_out,  // [B,U]
                __bf16*       __restrict__ hbf_out, // [B,U]
                int t) {
  const int lane = threadIdx.x & 31;
  const int mt   = threadIdx.x >> 5;   // m-tile (batch rows 16*mt .. 16*mt+15)
  const int nlo  = lane & 15;
  const int half = lane >> 4;          // 0: lanes 0-15, 1: lanes 16-31
  const int u0   = blockIdx.x * 16;
  const int u    = u0 + nlo;

  f32x8 acc_z  = {};
  f32x8 acc_r  = {};
  f32x8 acc_rh = {};
  f32x8 acc_xh = {};

  // A fragment row (ISA: lanes 0-15 and 16-31 both map to M = lane & 15)
  const int arow = mt * 16 + nlo;

  // --- recurrent GEMM: h[B,U] @ U[U,3U]  (K = 1024) -----------------------
  {
    const __bf16* aP  = hbf_in + (long)arow * kU;
    const __bf16* bzP = Ut + (long)(0 * kU + u) * kU;
    const __bf16* brP = Ut + (long)(1 * kU + u) * kU;
    const __bf16* bhP = Ut + (long)(2 * kU + u) * kU;
    for (int kk = 0; kk < kU; kk += 32) {
      const int ka = kk + 16 * half;   // A: K 0..15 / 16..31
      const int kb = kk + 16 * half;   // B: K 0..15 / 16..31
      __builtin_prefetch(bzP + kb + 128, 0, 1);   // global_prefetch_b8 path
      bf16x16 a  = make_frag(aP + ka, aP + ka + 8);
      bf16x16 bz = *reinterpret_cast<const bf16x16*>(bzP + kb);
      bf16x16 br = *reinterpret_cast<const bf16x16*>(brP + kb);
      bf16x16 bh = *reinterpret_cast<const bf16x16*>(bhP + kb);
      acc_z  = __builtin_amdgcn_wmma_f32_16x16x32_bf16(false, a, false, bz,
                                                       (short)0, acc_z, false, false);
      acc_r  = __builtin_amdgcn_wmma_f32_16x16x32_bf16(false, a, false, br,
                                                       (short)0, acc_r, false, false);
      acc_rh = __builtin_amdgcn_wmma_f32_16x16x32_bf16(false, a, false, bh,
                                                       (short)0, acc_rh, false, false);
    }
  }

  // --- input projection GEMM: xe[:,t,:] @ W[E,3U]  (K = 256) --------------
  {
    const __bf16* aP  = xe + ((long)arow * kT + t) * kE;
    const __bf16* bzP = Wt + (long)(0 * kU + u) * kE;
    const __bf16* brP = Wt + (long)(1 * kU + u) * kE;
    const __bf16* bhP = Wt + (long)(2 * kU + u) * kE;
    for (int kk = 0; kk < kE; kk += 32) {
      const int ka = kk + 16 * half;
      const int kb = kk + 16 * half;
      bf16x16 a  = make_frag(aP + ka, aP + ka + 8);
      bf16x16 bz = *reinterpret_cast<const bf16x16*>(bzP + kb);
      bf16x16 br = *reinterpret_cast<const bf16x16*>(brP + kb);
      bf16x16 bh = *reinterpret_cast<const bf16x16*>(bhP + kb);
      acc_z  = __builtin_amdgcn_wmma_f32_16x16x32_bf16(false, a, false, bz,
                                                       (short)0, acc_z, false, false);
      acc_r  = __builtin_amdgcn_wmma_f32_16x16x32_bf16(false, a, false, br,
                                                       (short)0, acc_r, false, false);
      acc_xh = __builtin_amdgcn_wmma_f32_16x16x32_bf16(false, a, false, bh,
                                                       (short)0, acc_xh, false, false);
    }
  }

  // --- gates + blend + mask ----------------------------------------------
  const float bz  = bias[u]           + bias[kN3 + u];            // b0_z + b1_z
  const float br  = bias[kU + u]      + bias[kN3 + kU + u];       // b0_r + b1_r
  const float bxh = bias[2 * kU + u];                             // b0_h
  const float brh = bias[kN3 + 2 * kU + u];                       // b1_h

#pragma unroll
  for (int v = 0; v < 8; ++v) {
    // C/D layout: VGPR v, lanes 0-15 -> M=v, lanes 16-31 -> M=v+8
    const int m = mt * 16 + v + 8 * half;            // batch row
    const float hold = hf_in[(long)m * kU + u];
    const int tok = x[m * kT + t];

    const float z  = fast_sigmoid(acc_z[v] + bz);
    const float r  = fast_sigmoid(acc_r[v] + br);
    const float hh = tanhf(acc_xh[v] + bxh + r * (acc_rh[v] + brh));
    float hnew = z * hold + (1.0f - z) * hh;
    hnew = (tok != 0) ? hnew : hold;                 // mask_zero carry-through

    out[((long)m * kT + t) * kU + u] = hnew;
    hf_out[(long)m * kU + u]  = hnew;
    hbf_out[(long)m * kU + u] = (__bf16)hnew;
  }
}

// ---------------------------------------------------------------------------
// Host driver
// ---------------------------------------------------------------------------

extern "C" void kernel_launch(void* const* d_in, const int* in_sizes, int n_in,
                              void* d_out, int out_size, void* d_ws, size_t ws_size,
                              hipStream_t stream) {
  (void)in_sizes; (void)n_in; (void)out_size; (void)ws_size;

  const int*   x      = (const int*)  d_in[0];
  const float* hidden = (const float*)d_in[1];
  const float* emb    = (const float*)d_in[2];
  const float* W      = (const float*)d_in[3];
  const float* Uk     = (const float*)d_in[4];
  const float* bias   = (const float*)d_in[5];
  float* out = (float*)d_out;

  char* ws = (char*)d_ws;
  size_t off = 0;
  auto carve = [&](size_t bytes) -> void* {
    void* p = ws + off;
    off += (bytes + 255) & ~(size_t)255;
    return p;
  };

  __bf16* Wt = (__bf16*)carve((size_t)kN3 * kE * sizeof(__bf16));  // 1.5 MB
  __bf16* Ut = (__bf16*)carve((size_t)kN3 * kU * sizeof(__bf16));  // 6.0 MB
  __bf16* xe = (__bf16*)carve((size_t)kB * kT * kE * sizeof(__bf16)); // 8 MB
  float*  hf[2];
  __bf16* hbf[2];
  hf[0]  = (float*)carve((size_t)kB * kU * sizeof(float));
  hf[1]  = (float*)carve((size_t)kB * kU * sizeof(float));
  hbf[0] = (__bf16*)carve((size_t)kB * kU * sizeof(__bf16));
  hbf[1] = (__bf16*)carve((size_t)kB * kU * sizeof(__bf16));

  // One-time prep (re-done every call: deterministic, graph-capture safe).
  convert_transpose_kernel<<<(kE * kN3) / 256, 256, 0, stream>>>(W,  Wt, kE, kN3);
  convert_transpose_kernel<<<(kU * kN3) / 256, 256, 0, stream>>>(Uk, Ut, kU, kN3);
  embed_bf16_kernel<<<(kB * kT * kE) / 256, 256, 0, stream>>>(x, emb, xe);
  init_h_kernel<<<(kB * kU) / 256, 256, 0, stream>>>(hidden, hf[0], hbf[0]);

  // Sequential GRU scan: ping-pong hidden state buffers.
  for (int t = 0; t < kT; ++t) {
    const int in = t & 1;
    const int o  = in ^ 1;
    gru_step_kernel<<<dim3(kU / 16), dim3(128), 0, stream>>>(
        x, bias, Ut, Wt, xe, hbf[in], hf[in], out, hf[o], hbf[o], t);
  }

  // t=255 is odd -> wrote hf[0]; that's the final state.
  copy_state_kernel<<<(kB * kU) / 256, 256, 0, stream>>>(
      hf[0], out + (size_t)kB * kT * kU);
}
